// MoELayer_22720376995909
// MI455X (gfx1250) — compile-verified, matrix operand order checked by reading, not run on hardware
//
#include <hip/hip_runtime.h>
#include <hip/hip_bf16.h>
#include <math.h>

#define D_MODEL 768
#define N_EXPERTS 16
#define D_FF 3072
#define TOKENS (4 * 2048)

#define TM 64            // tokens per FFN tile
#define KC 64            // d_ff chunk width
#define XS_STRIDE 776    // bf16 elems per x-tile / W1t row (768 + 8 pad)
#define HS_STRIDE 72     // bf16 elems per h-chunk / W2t row (64 + 8 pad)

// LDS layout (bytes)
#define XS_OFF   0
#define XS_BYTES (TM * XS_STRIDE * 2)              // 99328
#define HS_OFF   (XS_OFF + XS_BYTES)
#define HS_BYTES (TM * HS_STRIDE * 2)              // 9216
#define TOK_OFF  (HS_OFF + HS_BYTES)
#define TOK_BYTES (TM * 8)                         // 512
#define WB_OFF   (TOK_OFF + TOK_BYTES)             // 109056 (16B aligned)
#define WB_BYTES (D_MODEL * HS_STRIDE * 2)         // 110592 (fits W1t 99328 too)
#define SMEM_BYTES (WB_OFF + WB_BYTES)             // 219648

#define AS1 __attribute__((address_space(1)))
#define AS3 __attribute__((address_space(3)))

typedef __attribute__((ext_vector_type(16))) __bf16 v16bf;
typedef __attribute__((ext_vector_type(8)))  float  v8f;
typedef __attribute__((ext_vector_type(4)))  int    v4i;

__device__ __forceinline__ v8f v8f_zero() {
    v8f z;
#pragma unroll
    for (int i = 0; i < 8; ++i) z[i] = 0.0f;
    return z;
}

// Async global->LDS 16-byte copy (CDNA5; ASYNCcnt-tracked).
__device__ __forceinline__ void async_copy_b128(char* lds_dst, const void* gsrc) {
#if __has_builtin(__builtin_amdgcn_global_load_async_to_lds_b128)
    __builtin_amdgcn_global_load_async_to_lds_b128(
        (AS1 v4i*)gsrc, (AS3 v4i*)lds_dst, 0, 0);
#else
    unsigned loff = (unsigned)(__UINTPTR_TYPE__)(AS3 char*)lds_dst;
    asm volatile("global_load_async_to_lds_b128 %0, %1, off"
                 :: "v"(loff), "v"(gsrc) : "memory");
#endif
}

__device__ __forceinline__ void wait_async0() {
#if __has_builtin(__builtin_amdgcn_s_wait_asynccnt)
    __builtin_amdgcn_s_wait_asynccnt(0);
#else
    asm volatile("s_wait_asynccnt 0x0" ::: "memory");
#endif
}

// ---------------------------------------------------------------------------
// Router: 1 wave32 per token. 16 logits -> softmax -> top-2 -> bucket scatter.
// ---------------------------------------------------------------------------
__global__ __launch_bounds__(256) void moe_router(
    const float* __restrict__ x, const float* __restrict__ Wr,
    int* __restrict__ counts, int* __restrict__ btok, float* __restrict__ bgate)
{
    int wid  = threadIdx.x >> 5;
    int lane = threadIdx.x & 31;
    int t = blockIdx.x * 8 + wid;
    if (t >= TOKENS) return;
    const float* xt = x + (size_t)t * D_MODEL;

    float logits[N_EXPERTS];
#pragma unroll
    for (int e = 0; e < N_EXPERTS; ++e) {
        const float* wr = Wr + e * D_MODEL;
        float s = 0.0f;
        for (int j = lane; j < D_MODEL; j += 32) s += xt[j] * wr[j];
#pragma unroll
        for (int off = 16; off > 0; off >>= 1) s += __shfl_down(s, off, 32);
        logits[e] = s;   // valid on lane 0
    }
    if (lane == 0) {
        float m = logits[0];
#pragma unroll
        for (int e = 1; e < N_EXPERTS; ++e) m = fmaxf(m, logits[e]);
        float p[N_EXPERTS];
        float sum = 0.0f;
#pragma unroll
        for (int e = 0; e < N_EXPERTS; ++e) { p[e] = __expf(logits[e] - m); sum += p[e]; }
        float inv = 1.0f / sum;
        int i1 = 0; float p1 = -1.0f;
#pragma unroll
        for (int e = 0; e < N_EXPERTS; ++e) { if (p[e] > p1) { p1 = p[e]; i1 = e; } }
        int i2 = 0; float p2 = -1.0f;
#pragma unroll
        for (int e = 0; e < N_EXPERTS; ++e) { if (e != i1 && p[e] > p2) { p2 = p[e]; i2 = e; } }

        int pos1 = atomicAdd(&counts[i1], 1);
        btok [i1 * TOKENS + pos1] = t;
        bgate[i1 * TOKENS + pos1] = p1 * inv;
        int pos2 = atomicAdd(&counts[i2], 1);
        btok [i2 * TOKENS + pos2] = t;
        bgate[i2 * TOKENS + pos2] = p2 * inv;
    }
}

// ---------------------------------------------------------------------------
// out = x  (residual base; expert blocks atomically accumulate on top)
// ---------------------------------------------------------------------------
__global__ __launch_bounds__(256) void moe_copy_x(
    const float4* __restrict__ x, float4* __restrict__ out, int n)
{
    int i = blockIdx.x * blockDim.x + threadIdx.x;
    if (i < n) out[i] = x[i];
}

// ---------------------------------------------------------------------------
// Transpose + fp32->bf16 convert: in[e][R][C] -> out[e][C][R].
// 32x32 tiles via LDS; coalesced read and write.
// ---------------------------------------------------------------------------
__global__ __launch_bounds__(256) void transpose_cvt_bf16(
    const float* __restrict__ in, __bf16* __restrict__ out, int R, int C)
{
    __shared__ float tile[32][33];
    const int tx = threadIdx.x & 31;
    const int ty = threadIdx.x >> 5;           // 0..7
    const int r0 = blockIdx.y * 32;
    const int c0 = blockIdx.x * 32;
    const float* ine  = in  + (size_t)blockIdx.z * R * C;
    __bf16*      oute = out + (size_t)blockIdx.z * R * C;
#pragma unroll
    for (int i = 0; i < 4; ++i)
        tile[ty + i * 8][tx] = ine[(size_t)(r0 + ty + i * 8) * C + c0 + tx];
    __syncthreads();
#pragma unroll
    for (int i = 0; i < 4; ++i)
        oute[(size_t)(c0 + ty + i * 8) * R + r0 + tx] = (__bf16)tile[tx][ty + i * 8];
}

// ---------------------------------------------------------------------------
// Fused expert FFN. One block = (expert, 64-token tile). 512 thr = 16 wave32s.
// Weights pre-converted to bf16, transposed (K contiguous per output column):
//   w1b[e][f][k] (f<3072, k<768), w2b[e][n][k] (n<768, k<3072).
// Per d_ff chunk: async-stage W1t -> GEMM1(+GELU) -> async-stage W2t -> GEMM2.
// ---------------------------------------------------------------------------
__global__ __launch_bounds__(512) void moe_ffn(
    const float* __restrict__ x,
    const __bf16* __restrict__ w1b, const float* __restrict__ b1,
    const __bf16* __restrict__ w2b, const float* __restrict__ b2,
    const int* __restrict__ counts, const int* __restrict__ btok,
    const float* __restrict__ bgate, float* __restrict__ out)
{
    extern __shared__ char smem[];
    __bf16* xs   = (__bf16*)(smem + XS_OFF);   // [64][776]
    __bf16* hs   = (__bf16*)(smem + HS_OFF);   // [64][72]
    int*    ltok = (int*)  (smem + TOK_OFF);
    float*  lgat = (float*)(smem + TOK_OFF + TM * 4);
    __bf16* wb   = (__bf16*)(smem + WB_OFF);   // W1t [64][776] or W2t [768][72]

    const int e    = blockIdx.x;
    const int tile = blockIdx.y;
    const int cnt  = counts[e];
    const int row0 = tile * TM;
    if (row0 >= cnt) return;
    const int rows = min(TM, cnt - row0);

    const int tid   = threadIdx.x;
    const int lane  = tid & 31;
    const int wid   = tid >> 5;                 // 0..15
    const int rlane = lane & 15;
    const int half  = (lane >= 16) ? 8 : 0;     // K offset for A/B frag halves

    if (tid < TM) {
        if (tid < rows) {
            ltok[tid] = btok [e * TOKENS + row0 + tid];
            lgat[tid] = bgate[e * TOKENS + row0 + tid];
        } else {
            ltok[tid] = -1;
            lgat[tid] = 0.0f;
        }
    }
    __syncthreads();

    // gather x tile (fp32 -> bf16): 8 threads per row, 96 cols each
    {
        int r   = tid >> 3;
        int c0  = (tid & 7) * 96;
        int tok = ltok[r];
        __bf16* xr = xs + (size_t)r * XS_STRIDE + c0;
        if (tok >= 0) {
            const float4* xp = (const float4*)(x + (size_t)tok * D_MODEL + c0);
#pragma unroll
            for (int j = 0; j < 24; ++j) {
                float4 v = xp[j];
                xr[j * 4 + 0] = (__bf16)v.x;
                xr[j * 4 + 1] = (__bf16)v.y;
                xr[j * 4 + 2] = (__bf16)v.z;
                xr[j * 4 + 3] = (__bf16)v.w;
            }
        } else {
#pragma unroll
            for (int j = 0; j < 96; ++j) xr[j] = (__bf16)0.0f;
        }
    }
    __syncthreads();

    const __bf16* w1e = w1b + (size_t)e * D_FF * D_MODEL;
    const __bf16* w2e = w2b + (size_t)e * D_MODEL * D_FF;
    const float*  b1e = b1 + (size_t)e * D_FF;
    const float*  b2e = b2 + (size_t)e * D_MODEL;

    // GEMM1: one 16x16 h-tile per wave
    const int g1_mt = wid & 3;
    const int g1_nt = wid >> 2;
    // GEMM2: rows (wid&1)*32..+31, cols (wid>>1)*96..+95
    const int mhalf   = wid & 1;
    const int colbase = (wid >> 1) * 96;

    v8f acc[12];                                // acc[nt*2 + mi]
#pragma unroll
    for (int i = 0; i < 12; ++i) acc[i] = v8f_zero();

    for (int ff0 = 0; ff0 < D_FF; ff0 += KC) {
        // ---- async-stage W1t chunk: wb[f][k], f in [ff0,ff0+64), k in [0,768)
        {
#pragma unroll
            for (int j = 0; j < 12; ++j) {
                int seg = tid + j * 512;        // 0..6143, 16B each
                int row = seg / 96;             // local f (0..63)
                int off = (seg % 96) * 8;       // bf16 elems
                async_copy_b128(
                    (char*)(wb + (size_t)row * XS_STRIDE + off),
                    w1e + (size_t)(ff0 + row) * D_MODEL + off);
            }
            wait_async0();
        }
        __syncthreads();

        // ---- GEMM1: h[64,KC] = gelu(xs @ W1 chunk + b1), 1 tile per wave
        {
            v8f hacc = v8f_zero();
            const __bf16* xrow = xs + (size_t)(g1_mt * 16 + rlane) * XS_STRIDE;
            const __bf16* wrow = wb + (size_t)(g1_nt * 16 + rlane) * XS_STRIDE;
            for (int k0 = 0; k0 < D_MODEL; k0 += 32) {
                const int ab = k0 + half;
                v16bf a, b;
#pragma unroll
                for (int j = 0; j < 8; ++j) {
                    a[j]     = xrow[ab + j];
                    a[j + 8] = xrow[ab + 16 + j];
                    b[j]     = wrow[ab + j];
                    b[j + 8] = wrow[ab + 16 + j];
                }
                hacc = __builtin_amdgcn_wmma_f32_16x16x32_bf16(
                    false, a, false, b, (short)0, hacc, false, false);
            }
            const float bias = b1e[ff0 + g1_nt * 16 + rlane];
#pragma unroll
            for (int v = 0; v < 8; ++v) {
                float h = hacc[v] + bias;
                float g = 0.5f * h * (1.0f + erff(h * 0.70710678118f));
                hs[(size_t)(g1_mt * 16 + v + half) * HS_STRIDE
                   + g1_nt * 16 + rlane] = (__bf16)g;
            }
        }
        __syncthreads();

        // ---- async-stage W2t chunk: wb[n][kk], n in [0,768), kk in [0,64)
        {
#pragma unroll
            for (int j = 0; j < 12; ++j) {
                int seg = tid + j * 512;        // 0..6143, 16B each
                int row = seg >> 3;             // n (0..767)
                int off = (seg & 7) * 8;        // bf16 elems within chunk
                async_copy_b128(
                    (char*)(wb + (size_t)row * HS_STRIDE + off),
                    w2e + (size_t)row * D_FF + ff0 + off);
            }
            wait_async0();
        }
        __syncthreads();

        // ---- GEMM2: acc[32x96] += h[64xKC] @ W2 chunk
#pragma unroll
        for (int ks = 0; ks < 2; ++ks) {
            const int ab = ks * 32 + half;
            v16bf afr[2];
#pragma unroll
            for (int mi = 0; mi < 2; ++mi) {
                const __bf16* hrow =
                    hs + (size_t)(mhalf * 32 + mi * 16 + rlane) * HS_STRIDE;
#pragma unroll
                for (int j = 0; j < 8; ++j) {
                    afr[mi][j]     = hrow[ab + j];
                    afr[mi][j + 8] = hrow[ab + 16 + j];
                }
            }
#pragma unroll
            for (int nt = 0; nt < 6; ++nt) {
                const __bf16* wrow =
                    wb + (size_t)(colbase + nt * 16 + rlane) * HS_STRIDE;
                v16bf b;
#pragma unroll
                for (int j = 0; j < 8; ++j) {
                    b[j]     = wrow[ab + j];
                    b[j + 8] = wrow[ab + 16 + j];
                }
#pragma unroll
                for (int mi = 0; mi < 2; ++mi) {
                    acc[nt * 2 + mi] = __builtin_amdgcn_wmma_f32_16x16x32_bf16(
                        false, afr[mi], false, b, (short)0, acc[nt * 2 + mi],
                        false, false);
                }
            }
        }
        __syncthreads();
    }

    // ---- Epilogue: out[tok,col] += gate * (acc + b2[col])
#pragma unroll
    for (int nt = 0; nt < 6; ++nt) {
        const int col = colbase + nt * 16 + rlane;
        const float bb = b2e[col];
#pragma unroll
        for (int mi = 0; mi < 2; ++mi) {
            v8f a = acc[nt * 2 + mi];
#pragma unroll
            for (int v = 0; v < 8; ++v) {
                const int row = mhalf * 32 + mi * 16 + v + half;
                const int tok = ltok[row];
                if (tok >= 0) {
                    atomicAdd(&out[(size_t)tok * D_MODEL + col],
                              lgat[row] * (a[v] + bb));
                }
            }
        }
    }
}

// ---------------------------------------------------------------------------
extern "C" void kernel_launch(void* const* d_in, const int* in_sizes, int n_in,
                              void* d_out, int out_size, void* d_ws, size_t ws_size,
                              hipStream_t stream) {
    (void)in_sizes; (void)n_in; (void)out_size; (void)ws_size;
    const float* x  = (const float*)d_in[0];
    const float* Wr = (const float*)d_in[1];
    const float* W1 = (const float*)d_in[2];
    const float* b1 = (const float*)d_in[3];
    const float* W2 = (const float*)d_in[4];
    const float* b2 = (const float*)d_in[5];
    float* out = (float*)d_out;

    // ws layout: counts | btok | bgate | w1b bf16 (transposed) | w2b bf16
    char*   ws     = (char*)d_ws;
    int*    counts = (int*)ws;
    int*    btok   = (int*)  (ws + 256);
    float*  bgate  = (float*)(ws + 256 + (size_t)N_EXPERTS * TOKENS * 4);
    __bf16* w1b    = (__bf16*)(ws + 1048832);                    // [E][3072][768]
    __bf16* w2b    = (__bf16*)(ws + 1048832 + (size_t)N_EXPERTS * D_FF * D_MODEL * 2);

    (void)hipMemsetAsync(counts, 0, 256, stream);

    moe_router<<<TOKENS / 8, 256, 0, stream>>>(x, Wr, counts, btok, bgate);

    // Weight convert+transpose (one-time per launch, ~450MB traffic total)
    {
        dim3 g1(D_FF / 32, D_MODEL / 32, N_EXPERTS);   // W1 [768][3072] -> [3072][768]
        transpose_cvt_bf16<<<g1, 256, 0, stream>>>(W1, w1b, D_MODEL, D_FF);
        dim3 g2(D_MODEL / 32, D_FF / 32, N_EXPERTS);   // W2 [3072][768] -> [768][3072]
        transpose_cvt_bf16<<<g2, 256, 0, stream>>>(W2, w2b, D_FF, D_MODEL);
    }

    const int ncopy = TOKENS * D_MODEL / 4;
    moe_copy_x<<<(ncopy + 255) / 256, 256, 0, stream>>>(
        (const float4*)x, (float4*)out, ncopy);

    dim3 grid(N_EXPERTS, TOKENS / TM);   // tiles past count[e] exit early
    moe_ffn<<<grid, 512, SMEM_BYTES, stream>>>(x, w1b, b1, w2b, b2,
                                               counts, btok, bgate, out);
}
